// QKNormAttn_75488345194584
// MI455X (gfx1250) — matrix operand order repaired and checked by reading, not compiled, not run on hardware
//
#include <hip/hip_runtime.h>

#define M_TOK 2048
#define NH    32
#define HD    128
#define PCACHE 2048
#define NDIM  4096   // NH*HD
#define PTOT  4096   // PCACHE + M_TOK

typedef __attribute__((ext_vector_type(16))) __bf16       v16bf;
typedef __attribute__((ext_vector_type(2)))  __bf16       v2bf;
typedef __attribute__((ext_vector_type(8)))  float        v8f;
typedef __attribute__((ext_vector_type(4)))  unsigned int u32x4;
typedef __attribute__((ext_vector_type(4)))  float        f32x4;

union Frag { v16bf v; u32x4 u[2]; };

__device__ __forceinline__ unsigned short f2bf(float x) {
  unsigned int u = __float_as_uint(x);
  u += 0x7fffu + ((u >> 16) & 1u);          // round-to-nearest-even
  return (unsigned short)(u >> 16);
}

#if __has_builtin(__builtin_amdgcn_cvt_pk_bf16_f32)
__device__ __forceinline__ unsigned int pack2(float a, float b) {
  union { v2bf v; unsigned int u; } cv;
  cv.v = __builtin_amdgcn_cvt_pk_bf16_f32(a, b);   // v_cvt_pk_bf16_f32
  return cv.u;
}
#else
__device__ __forceinline__ unsigned int pack2(float a, float b) {
  return (unsigned int)f2bf(a) | ((unsigned int)f2bf(b) << 16);
}
#endif

// Async global -> LDS 128-bit copy (ASYNCcnt-tracked), GV addressing mode.
__device__ __forceinline__ void async_b128(unsigned lds_off, const void* gaddr) {
  asm volatile("global_load_async_to_lds_b128 %0, %1, off"
               :: "v"(lds_off), "v"((unsigned long long)(size_t)gaddr)
               : "memory");
}
__device__ __forceinline__ void wait_async_le4() {
  asm volatile("s_wait_asynccnt 0x4" ::: "memory");
}
__device__ __forceinline__ void wait_async_0() {
  asm volatile("s_wait_asynccnt 0x0" ::: "memory");
}
__device__ __forceinline__ unsigned lds_addr(const void* p) {
  return (unsigned)(size_t)p;   // flat LDS aperture keeps offset in low 32 bits
}

// ---------------------------------------------------------------------------
// Kernel 1: plain fp32 -> bf16 convert (X and the three weight matrices).
// ---------------------------------------------------------------------------
__global__ __launch_bounds__(256) void cvt_mat_kernel(
    const float* __restrict__ src, unsigned short* __restrict__ dst)
{
  size_t i = ((size_t)blockIdx.x * 256 + threadIdx.x) * 4;
  f32x4 a = *(const f32x4*)&src[i];
  *(unsigned int*)&dst[i]     = pack2(a.x, a.y);
  *(unsigned int*)&dst[i + 2] = pack2(a.z, a.w);
}

// ---------------------------------------------------------------------------
// Kernel 2: KV-cache fp32 -> bf16. K goes to (H, PTOT, D) rows [0,P);
// V goes TRANSPOSED to (H, D, PTOT) cols [0,P) so attention B-fragments of V
// are contiguous and need no per-chunk transpose.
// ---------------------------------------------------------------------------
__global__ __launch_bounds__(256) void cvt_cache_kernel(
    const float* __restrict__ cK, const float* __restrict__ cV,
    unsigned short* __restrict__ Kc, unsigned short* __restrict__ VcT)
{
  size_t i = ((size_t)blockIdx.x * 256 + threadIdx.x) * 4;   // float index
  const size_t HP = (size_t)PCACHE * HD;
  size_t h = i / HP, rem = i % HP;
  // K: same (p, d) layout, row offset 0
  size_t dk = h * ((size_t)PTOT * HD) + rem;
  f32x4 a = *(const f32x4*)&cK[i];
  *(unsigned int*)&Kc[dk]     = pack2(a.x, a.y);
  *(unsigned int*)&Kc[dk + 2] = pack2(a.z, a.w);
  // V: transpose (p, d) -> (d, p)
  size_t p = rem / HD, d = rem % HD;
  f32x4 b = *(const f32x4*)&cV[i];
  unsigned short* vp = VcT + h * ((size_t)HD * PTOT);
  vp[(d + 0) * PTOT + p] = f2bf(b.x);
  vp[(d + 1) * PTOT + p] = f2bf(b.y);
  vp[(d + 2) * PTOT + p] = f2bf(b.z);
  vp[(d + 3) * PTOT + p] = f2bf(b.w);
}

// ---------------------------------------------------------------------------
// Kernel 3: QKV projection GEMM y = X @ W^T (TN), bf16 in / bf16 WMMA,
// double-buffered async global->LDS tile loads, RMS-norm fused epilogue.
// blockIdx.z: 0 = Q (normed), 1 = K (normed, cache rows P..), 2 = V (transposed).
// ---------------------------------------------------------------------------
__global__ __launch_bounds__(256) void qkv_gemm_kernel(
    const unsigned short* __restrict__ Xb,
    const unsigned short* __restrict__ Wb,
    unsigned short* __restrict__ Qb, unsigned short* __restrict__ Kc,
    unsigned short* __restrict__ VcT)
{
  __shared__ union {
    unsigned short tiles[2][2][128][32];   // [buf][0=X,1=W][row][k]  32 KB
    float c[128][128];                     // epilogue bounce          64 KB
  } sm;
  __shared__ float rowscale[128];

  const int tid  = threadIdx.x;
  const int lane = tid & 31;
  const int w    = tid >> 5;
  const int wm   = w & 3, wn = w >> 2;     // 4 waves along M, 2 along N
  const int lm   = lane & 15, lh = lane >> 4;

  const int m0 = blockIdx.x * 128;
  const int hN = blockIdx.y;               // head index == N-block
  const int n0 = hN * 128;
  const int z  = blockIdx.z;
  const unsigned short* Wp = Wb + (size_t)z * NDIM * NDIM;

  const unsigned lbase = lds_addr(&sm.tiles[0][0][0][0]);

  auto issue_tile = [&](int buf, int k0) {
    #pragma unroll
    for (int i = 0; i < 2; i++) {
      int idx = tid + i * 256;             // 0..511
      int r   = idx >> 2;                  // 0..127
      int s   = (idx & 3) << 3;            // ushort offset 0,8,16,24
      unsigned lx = lbase + (unsigned)(buf * 16384 + r * 64 + s * 2);
      async_b128(lx,        Xb + (size_t)(m0 + r) * NDIM + k0 + s);
      async_b128(lx + 8192, Wp + (size_t)(n0 + r) * NDIM + k0 + s);
    }
  };

  const v8f vz = {0.f,0.f,0.f,0.f,0.f,0.f,0.f,0.f};
  v8f acc[2][4];
  #pragma unroll
  for (int i = 0; i < 2; i++)
    #pragma unroll
    for (int j = 0; j < 4; j++) acc[i][j] = vz;

  const int NSTEP = NDIM / 32;
  issue_tile(0, 0);
  for (int it = 0; it < NSTEP; ++it) {
    const int cur = it & 1;
    if (it + 1 < NSTEP) { issue_tile(1 - cur, (it + 1) * 32); wait_async_le4(); }
    else                { wait_async_0(); }
    __syncthreads();

    const unsigned short (*xs)[32] = sm.tiles[cur][0];
    const unsigned short (*ws)[32] = sm.tiles[cur][1];
    Frag a[2], b[4];
    #pragma unroll
    for (int mt = 0; mt < 2; mt++) {       // A: lane&15 = m, K runs by lane>>4
      int r = wm * 32 + mt * 16 + lm;
      a[mt].u[0] = *(const u32x4*)&xs[r][lh * 16];
      a[mt].u[1] = *(const u32x4*)&xs[r][lh * 16 + 8];
    }
    #pragma unroll
    for (int nt = 0; nt < 4; nt++) {       // B: lane&15 = n, 16 contiguous K
      int cc = wn * 64 + nt * 16 + lm;
      b[nt].u[0] = *(const u32x4*)&ws[cc][lh * 16];
      b[nt].u[1] = *(const u32x4*)&ws[cc][lh * 16 + 8];
    }
    #pragma unroll
    for (int mt = 0; mt < 2; mt++)
      #pragma unroll
      for (int nt = 0; nt < 4; nt++)
        acc[mt][nt] = __builtin_amdgcn_wmma_f32_16x16x32_bf16(
            false, a[mt].v, false, b[nt].v, (short)0, acc[mt][nt], false, false);
    __syncthreads();
  }

  // ---- epilogue: bounce C through LDS ----
  #pragma unroll
  for (int mt = 0; mt < 2; mt++)
    #pragma unroll
    for (int nt = 0; nt < 4; nt++)
      #pragma unroll
      for (int j = 0; j < 8; j++) {
        int r  = wm * 32 + mt * 16 + j + lh * 8;
        int cc = wn * 64 + nt * 16 + lm;
        sm.c[r][cc] = acc[mt][nt][j];
      }
  __syncthreads();

  if (z == 2) {
    // V: write transposed into (H, D, PTOT), cols [P+m0, P+m0+128)
    int d  = tid >> 1;
    int mh = (tid & 1) * 64;
    unsigned short* dcol =
        VcT + ((size_t)(hN * HD + d)) * PTOT + PCACHE + m0 + mh;
    #pragma unroll
    for (int mm = 0; mm < 64; mm += 2)
      *(unsigned int*)&dcol[mm] = pack2(sm.c[mh + mm][d], sm.c[mh + mm + 1][d]);
    return;
  }

  // Q / K: fused RMS-norm over the head dim (BN == HD == 128)
  if (tid < 128) {
    float s = 0.f;
    #pragma unroll 8
    for (int c = 0; c < 128; c++) { float v = sm.c[tid][c]; s += v * v; }
    rowscale[tid] = rsqrtf(s * (1.0f / 128.0f));
  }
  __syncthreads();

  unsigned short* dst; size_t rbase;
  if (z == 0) { dst = Qb; rbase = (size_t)hN * M_TOK + m0; }
  else        { dst = Kc; rbase = (size_t)hN * PTOT + PCACHE + m0; }

  int r  = tid >> 1;
  int c0 = (tid & 1) * 64;
  float sc = rowscale[r];
  unsigned short* drow = dst + (rbase + r) * HD;
  #pragma unroll
  for (int c = 0; c < 64; c += 2)
    *(unsigned int*)&drow[c0 + c] =
        pack2(sm.c[r][c0 + c] * sc, sm.c[r][c0 + c + 1] * sc);
}

// ---------------------------------------------------------------------------
// Kernel 4: streaming attention, unstabilized softmax, double-buffered async
// K/V chunk loads shared by all 8 waves. Block = (head, 128 q rows).
// ---------------------------------------------------------------------------
__global__ __launch_bounds__(256) void attn_kernel(
    const unsigned short* __restrict__ Qb,
    const unsigned short* __restrict__ Kc,
    const unsigned short* __restrict__ VcT,
    float* __restrict__ out)
{
  __shared__ unsigned short Ks[2][32][128];   // K chunk (p, d)        16 KB
  __shared__ unsigned short Vt[2][128][32];   // V chunk (d, p)        16 KB
  __shared__ unsigned short Sb[8][16][32];    // per-wave exp(S)        8 KB

  const int tid  = threadIdx.x;
  const int lane = tid & 31;
  const int w    = tid >> 5;
  const int lm   = lane & 15, lh = lane >> 4;
  const int h    = blockIdx.x;
  const int m0   = blockIdx.y * 128 + w * 16;

  const unsigned short* Kh = Kc  + (size_t)h * PTOT * HD;   // (p, d)
  const unsigned short* Vh = VcT + (size_t)h * HD * PTOT;   // (d, p)
  const unsigned kbase = lds_addr(&Ks[0][0][0]);
  const unsigned vbase = lds_addr(&Vt[0][0][0]);

  auto issue_chunk = [&](int buf, int pc) {
    #pragma unroll
    for (int i = 0; i < 2; i++) {
      int idx = tid + i * 256;               // 0..511
      // K chunk: fully contiguous 8 KB
      async_b128(kbase + (unsigned)(buf * 8192 + idx * 16),
                 Kh + (size_t)pc * HD + idx * 8);
      // V chunk: 128 rows of 64 B from the pre-transposed cache
      int r = idx >> 2, s = (idx & 3) << 3;
      async_b128(vbase + (unsigned)(buf * 8192 + r * 64 + s * 2),
                 Vh + (size_t)r * PTOT + pc + s);
    }
  };

  // Q fragments (16 rows x 128 d) resident for the whole key loop.
  Frag qa[4];
  const unsigned short* qrow = Qb + ((size_t)h * M_TOK + m0 + lm) * HD;
  #pragma unroll
  for (int f = 0; f < 4; f++) {
    qa[f].u[0] = *(const u32x4*)&qrow[f * 32 + lh * 16];
    qa[f].u[1] = *(const u32x4*)&qrow[f * 32 + lh * 16 + 8];
  }

  const v8f vz = {0.f,0.f,0.f,0.f,0.f,0.f,0.f,0.f};
  v8f o[8];
  #pragma unroll
  for (int i = 0; i < 8; i++) o[i] = vz;
  float rsum[8];
  #pragma unroll
  for (int j = 0; j < 8; j++) rsum[j] = 0.f;

  const int NCH = PTOT / 32;
  issue_chunk(0, 0);
  for (int it = 0; it < NCH; ++it) {
    const int cur = it & 1;
    if (it + 1 < NCH) { issue_chunk(1 - cur, (it + 1) * 32); wait_async_le4(); }
    else              { wait_async_0(); }
    __syncthreads();

    const unsigned short (*Kcur)[128] = Ks[cur];
    const unsigned short (*Vcur)[32]  = Vt[cur];

    // S = Q . K^T : two 16x16 score tiles (p 0..15, 16..31), contract d=128
    v8f s0 = vz, s1 = vz;
    #pragma unroll
    for (int f = 0; f < 4; f++) {
      Frag kb0, kb1;
      kb0.u[0] = *(const u32x4*)&Kcur[lm     ][f * 32 + lh * 16];
      kb0.u[1] = *(const u32x4*)&Kcur[lm     ][f * 32 + lh * 16 + 8];
      kb1.u[0] = *(const u32x4*)&Kcur[16 + lm][f * 32 + lh * 16];
      kb1.u[1] = *(const u32x4*)&Kcur[16 + lm][f * 32 + lh * 16 + 8];
      s0 = __builtin_amdgcn_wmma_f32_16x16x32_bf16(false, qa[f].v, false, kb0.v,
                                                   (short)0, s0, false, false);
      s1 = __builtin_amdgcn_wmma_f32_16x16x32_bf16(false, qa[f].v, false, kb1.v,
                                                   (short)0, s1, false, false);
    }

    // exp, row-sum accumulate, C->A re-layout via wave-private LDS tile
    #pragma unroll
    for (int j = 0; j < 8; j++) {
      float e0 = __expf(s0[j]);
      float e1 = __expf(s1[j]);
      rsum[j] += e0 + e1;
      unsigned int pk = pack2(e0, e1);
      int m = j + lh * 8;
      Sb[w][m][lm]      = (unsigned short)pk;
      Sb[w][m][16 + lm] = (unsigned short)(pk >> 16);
    }
    Frag af;
    af.u[0] = *(const u32x4*)&Sb[w][lm][lh * 16];
    af.u[1] = *(const u32x4*)&Sb[w][lm][lh * 16 + 8];

    // O += exp(S) . V : contract p=32, 8 output d-tiles
    #pragma unroll
    for (int dt = 0; dt < 8; dt++) {
      Frag vb;
      int cc = dt * 16 + lm;
      vb.u[0] = *(const u32x4*)&Vcur[cc][lh * 16];
      vb.u[1] = *(const u32x4*)&Vcur[cc][lh * 16 + 8];
      o[dt] = __builtin_amdgcn_wmma_f32_16x16x32_bf16(false, af.v, false, vb.v,
                                                      (short)0, o[dt], false, false);
    }
    __syncthreads();
  }

  // Reduce row sums across the 16-lane half (matches C-layout m mapping)
  #pragma unroll
  for (int j = 0; j < 8; j++)
    #pragma unroll
    for (int off = 1; off < 16; off <<= 1)
      rsum[j] += __shfl_xor(rsum[j], off, 32);

  // Normalize and store fp32 output (M, N) with per-head column offset
  #pragma unroll
  for (int dt = 0; dt < 8; dt++)
    #pragma unroll
    for (int j = 0; j < 8; j++) {
      int mrow = m0 + j + lh * 8;
      int dcol = dt * 16 + lm;
      out[(size_t)mrow * NDIM + h * HD + dcol] = o[dt][j] / rsum[j];
    }
}

// ---------------------------------------------------------------------------
extern "C" void kernel_launch(void* const* d_in, const int* in_sizes, int n_in,
                              void* d_out, int out_size, void* d_ws, size_t ws_size,
                              hipStream_t stream) {
  const float* X  = (const float*)d_in[0];
  const float* Wq = (const float*)d_in[1];
  const float* Wk = (const float*)d_in[2];
  const float* Wv = (const float*)d_in[3];
  const float* cK = (const float*)d_in[4];
  const float* cV = (const float*)d_in[5];
  float* out = (float*)d_out;

  const size_t MN = (size_t)M_TOK * NDIM;      // 8M elems
  const size_t NN = (size_t)NDIM * NDIM;       // 16M elems
  // Workspace (bf16): Xb 16MB | Wb 96MB | Qn 16MB | Kcache 32MB | VcacheT 32MB
  unsigned short* Xb = (unsigned short*)d_ws;
  unsigned short* Wb = Xb + MN;
  unsigned short* Qb = Wb + 3 * NN;
  unsigned short* Kc = Qb + (size_t)NH * M_TOK * HD;
  unsigned short* Vc = Kc + (size_t)NH * PTOT * HD;

  cvt_mat_kernel<<<dim3(MN / 1024), 256, 0, stream>>>(X,  Xb);
  cvt_mat_kernel<<<dim3(NN / 1024), 256, 0, stream>>>(Wq, Wb);
  cvt_mat_kernel<<<dim3(NN / 1024), 256, 0, stream>>>(Wk, Wb + NN);
  cvt_mat_kernel<<<dim3(NN / 1024), 256, 0, stream>>>(Wv, Wb + 2 * NN);
  cvt_cache_kernel<<<dim3((NH * PCACHE * HD) / 1024), 256, 0, stream>>>(
      cK, cV, Kc, Vc);

  dim3 g1(M_TOK / 128, NDIM / 128, 3);
  qkv_gemm_kernel<<<g1, 256, 0, stream>>>(Xb, Wb, Qb, Kc, Vc);

  dim3 g2(NH, M_TOK / 128, 1);
  attn_kernel<<<g2, 256, 0, stream>>>(Qb, Kc, Vc, out);
}
